// GraphCNNGANG_15857019256866
// MI455X (gfx1250) — matrix operand order repaired and checked by reading, not compiled, not run on hardware
//
#include <hip/hip_runtime.h>
#include <hip/hip_bf16.h>

#define ALPHA 0.2f
#define EPS 1e-5f
#define BATCH 1024
#define NHITS 100
#define NODES (BATCH * NHITS)   // 102400
#define STAT_WAVES 1024         // waves used for BN partial sums

typedef __attribute__((ext_vector_type(16))) _Float16 v16h;
typedef __attribute__((ext_vector_type(8)))  _Float16 v8h;
typedef __attribute__((ext_vector_type(8)))  float    v8f;

static __device__ __forceinline__ float leaky(float v) { return v > 0.f ? v : ALPHA * v; }

static __device__ __forceinline__ v16h cat8(v8h lo, v8h hi) {
    return __builtin_shufflevector(lo, hi, 0, 1, 2, 3, 4, 5, 6, 7, 8, 9, 10, 11, 12, 13, 14, 15);
}

// ---------------------------------------------------------------------------
// f32 -> f16 conversion (row-major A operand for the dense GEMM)
// ---------------------------------------------------------------------------
__global__ void cvt_f16_kernel(const float* __restrict__ in, _Float16* __restrict__ out, int n) {
    int t = blockIdx.x * blockDim.x + threadIdx.x;
    if (t < n) out[t] = (_Float16)in[t];
}

// ---------------------------------------------------------------------------
// Repack W_dense (128 x 6400 f32) into WMMA-B fragment-major f16:
// each lane's 16 B-halves become one contiguous 32-byte run.
// ---------------------------------------------------------------------------
__global__ void build_bdense_kernel(const float* __restrict__ W, _Float16* __restrict__ Bp) {
    int t = blockIdx.x * blockDim.x + threadIdx.x;
    if (t >= 128 * 6400) return;
    int i = t & 15;
    int L = (t >> 4) & 31;
    int q = t >> 9;             // 0..1599
    int nt = q % 400;
    int slab = q / 400;
    int k = slab * 32 + (L >> 4) * 8 + (i & 7) + ((i >> 3) << 4);
    int col = nt * 16 + (L & 15);
    Bp[t] = (_Float16)W[(size_t)k * 6400 + col];
}

// Wflat1 packed: source We0 (64 x 2048), logical B = Wflat[(i*64+k), o] = We0[k, i*32+o]
__global__ void build_wflat1_kernel(const float* __restrict__ We, _Float16* __restrict__ Wf) {
    int t = blockIdx.x * blockDim.x + threadIdx.x;
    if (t >= 4096 * 32) return;
    int i = t & 15;
    int L = (t >> 4) & 31;
    int q = t >> 9;
    int tcol = q & 1;
    int slab = q >> 1;          // 0..127
    int kflat = slab * 32 + (L >> 4) * 8 + (i & 7) + ((i >> 3) << 4);
    int o = tcol * 16 + (L & 15);
    int ii = kflat >> 6;
    int kk = kflat & 63;
    Wf[t] = (_Float16)We[kk * 2048 + ii * 32 + o];
}

// Wflat2 packed: source We1 (32 x 96), logical B = Wflat[(i*32+k), o<3] = We1[k, i*3+o], else 0
__global__ void build_wflat2_kernel(const float* __restrict__ We, _Float16* __restrict__ Wf) {
    int t = blockIdx.x * blockDim.x + threadIdx.x;
    if (t >= 1024 * 16) return;
    int i = t & 15;
    int L = (t >> 4) & 31;
    int slab = t >> 9;          // 0..31
    int kflat = slab * 32 + (L >> 4) * 8 + (i & 7) + ((i >> 3) << 4);
    int o = L & 15;
    int ii = kflat >> 5;
    int kk = kflat & 31;
    Wf[t] = (o < 3) ? (_Float16)We[kk * 96 + ii * 3 + o] : (_Float16)0.0f;
}

// ---------------------------------------------------------------------------
// Dense layer: C(1024x6400) = leaky(A(1024x128) @ B(128x6400) + bias)
// One wave per (16-row, 64-col) group: 4 accumulators, 16 WMMAs, A amortized 4x.
// ---------------------------------------------------------------------------
__global__ void __launch_bounds__(256)
dense_wmma_kernel(const _Float16* __restrict__ A, const _Float16* __restrict__ Bp,
                  const float* __restrict__ bias, float* __restrict__ C) {
    const int wave = threadIdx.x >> 5;
    const int lane = threadIdx.x & 31;
    const int wg = blockIdx.x * 8 + wave;     // 0..6399
    const int mt = wg / 100;
    const int ng = wg % 100;                  // group of 4 col-tiles
    const int l16 = lane & 15, hlf = lane >> 4;
    const int mrow = mt * 16 + l16;
    v8f c[4] = {};
#pragma unroll
    for (int slab = 0; slab < 4; ++slab) {
        const int kk = slab * 32;
        v8h lo = *(const v8h*)(A + (size_t)mrow * 128 + kk + 8 * hlf);
        v8h hi = *(const v8h*)(A + (size_t)mrow * 128 + kk + 16 + 8 * hlf);
        v16h a = cat8(lo, hi);
#pragma unroll
        for (int t = 0; t < 4; ++t) {
            v16h b = *(const v16h*)(Bp + (((size_t)slab * 400 + ng * 4 + t) * 32 + lane) * 16);
            c[t] = __builtin_amdgcn_wmma_f32_16x16x32_f16(false, a, false, b, (short)0, c[t], false, false);
        }
    }
#pragma unroll
    for (int t = 0; t < 4; ++t) {
        int col = ng * 64 + t * 16 + l16;
        float bv = bias[col];
#pragma unroll
        for (int r = 0; r < 8; ++r) {
            int m = mt * 16 + r + hlf * 8;
            C[(size_t)m * 6400 + col] = leaky(c[t][r] + bv);
        }
    }
}

// ---------------------------------------------------------------------------
// kNN (k=1) per graph: block per batch, nodes in LDS, reference distance formula.
// ---------------------------------------------------------------------------
template <int DIN>
__global__ void knn1_kernel(const float* __restrict__ h, int* __restrict__ idx) {
    __shared__ float sx[NHITS * DIN];
    __shared__ float ssq[NHITS];
    const int b = blockIdx.x;
    const float4* hb4 = (const float4*)(h + (size_t)b * NHITS * DIN);
    float4* sx4 = (float4*)sx;
    for (int t = threadIdx.x; t < NHITS * DIN / 4; t += blockDim.x) sx4[t] = hb4[t];
    __syncthreads();
    for (int n = threadIdx.x; n < NHITS; n += blockDim.x) {
        float s = 0.f;
        for (int d = 0; d < DIN; ++d) { float v = sx[n * DIN + d]; s += v * v; }
        ssq[n] = s;
    }
    __syncthreads();
    for (int n = threadIdx.x; n < NHITS; n += blockDim.x) {
        float best = 3.4e38f; int bi = 0;
        for (int m = 0; m < NHITS; ++m) {
            if (m == n) continue;
            float dot = 0.f;
            for (int d = 0; d < DIN; ++d) dot += sx[n * DIN + d] * sx[m * DIN + d];
            float d2 = ssq[n] + ssq[m] - 2.f * dot;
            if (d2 < best) { best = d2; bi = m; }
        }
        idx[b * NHITS + n] = bi;
    }
}

// ---------------------------------------------------------------------------
// Gather neighbor features (float4 vectorized)
// ---------------------------------------------------------------------------
template <int DIN>
__global__ void gather_kernel(const float* __restrict__ h, const int* __restrict__ idx,
                              float* __restrict__ xj) {
    constexpr int D4 = DIN / 4;
    int t = blockIdx.x * blockDim.x + threadIdx.x;
    if (t >= NODES * D4) return;
    int d = t % D4, nid = t / D4;
    int b = nid / NHITS;
    ((float4*)xj)[(size_t)nid * D4 + d] =
        ((const float4*)h)[((size_t)b * NHITS + idx[nid]) * D4 + d];
}

// ---------------------------------------------------------------------------
// base[nid,o] = rb[o] + sum_i h[nid,i]*root[i,o] + sum_i xj[nid,i]*be[i*DOUT+o]
// (written into the pre-BN buffer; the WMMA msg kernel accumulates on top)
// ---------------------------------------------------------------------------
template <int DIN, int DOUT, int STR>
__global__ void base_kernel(const float* __restrict__ h, const float* __restrict__ xj,
                            const float* __restrict__ root, const float* __restrict__ rb,
                            const float* __restrict__ be, float* __restrict__ pre) {
    int t = blockIdx.x * blockDim.x + threadIdx.x;
    if (t >= NODES * STR) return;
    int o = t % STR, nid = t / STR;
    float acc = 0.f;
    if (o < DOUT) {
        acc = rb[o];
        const float* hr = h + (size_t)nid * DIN;
        const float* xr = xj + (size_t)nid * DIN;
        for (int i = 0; i < DIN; ++i) acc += hr[i] * root[i * DOUT + o];
        for (int i = 0; i < DIN; ++i) acc += xr[i] * be[i * DOUT + o];
    }
    pre[t] = acc;
}

// ---------------------------------------------------------------------------
// Fused NNConv message: pre += P @ Wflat, P[n, i*DIN+k] = xj[n,i]*eattr[n,k],
// generated on the fly from LDS-staged 16-node tiles (eattr pre-differenced
// at staging; inner loop reads float2 pairs -> ds_load_b64).
// ---------------------------------------------------------------------------
template <int DIN, int NT>
__global__ void __launch_bounds__(256)
msg_wmma_kernel(const float* __restrict__ h, const float* __restrict__ xj,
                const _Float16* __restrict__ Wfp, float* __restrict__ pre) {
    constexpr int STR = NT * 16;
    constexpr int SLABS = DIN * DIN / 32;
    __shared__ float sXj[8][16][DIN];
    __shared__ float sE[8][16][DIN];    // eattr = xj - x
    const int wave = threadIdx.x >> 5;
    const int lane = threadIdx.x & 31;
    const int tile = blockIdx.x * 8 + wave;
    const int row0 = tile * 16;

    {   // vectorized staging; difference computed once here, not in the K loop
        const float4* xj4 = (const float4*)(xj + (size_t)row0 * DIN);
        const float4* h4  = (const float4*)(h + (size_t)row0 * DIN);
        float4* sXj4 = (float4*)&sXj[wave][0][0];
        float4* sE4  = (float4*)&sE[wave][0][0];
        for (int t = lane; t < 16 * DIN / 4; t += 32) {
            float4 a = xj4[t], b = h4[t], e;
            e.x = a.x - b.x; e.y = a.y - b.y; e.z = a.z - b.z; e.w = a.w - b.w;
            sXj4[t] = a; sE4[t] = e;
        }
    }
    __syncthreads();

    const int m   = lane & 15;   // A-row for this lane; also C/D column index
    const int hlf = lane >> 4;
    v8f c[NT];
#pragma unroll
    for (int t = 0; t < NT; ++t)
#pragma unroll
        for (int r = 0; r < 8; ++r)
            c[t][r] = pre[(size_t)(row0 + r + hlf * 8) * STR + t * 16 + m];

    for (int slab = 0; slab < SLABS; ++slab) {
        const int kk = slab * 32;
        v16h a;
#pragma unroll
        for (int r8 = 0; r8 < 8; ++r8) {
            // even-aligned K pair per A-VGPR (ISA 16-bit A layout)
            int koff = ((r8 & 4) ? 16 : 0) + hlf * 8 + ((r8 & 3) << 1);
            int kidx = kk + koff;                 // even
            int ii = kidx / DIN;                  // pair-invariant (bit0 unused)
            int k2 = kidx % DIN;                  // even -> float2-aligned
            float xv = sXj[wave][m][ii];
            float2 e2 = *(const float2*)&sE[wave][m][k2];   // ds_load_b64
            a[2 * r8]     = (_Float16)(xv * e2.x);
            a[2 * r8 + 1] = (_Float16)(xv * e2.y);
        }
#pragma unroll
        for (int t = 0; t < NT; ++t) {
            v16h b = *(const v16h*)(Wfp + (((size_t)slab * NT + t) * 32 + lane) * 16);
            c[t] = __builtin_amdgcn_wmma_f32_16x16x32_f16(false, a, false, b, (short)0, c[t], false, false);
        }
    }
#pragma unroll
    for (int t = 0; t < NT; ++t)
#pragma unroll
        for (int r = 0; r < 8; ++r)
            pre[(size_t)(row0 + r + hlf * 8) * STR + t * 16 + m] = c[t][r];
}

// ---------------------------------------------------------------------------
// BatchNorm statistics, coalesced: lane -> feature column, wave streams whole
// rows; per-wave partials reduced by a fixed-order kernel (deterministic).
// ---------------------------------------------------------------------------
template <int STR>
__global__ void __launch_bounds__(256)
bn_stats_part_kernel(const float* __restrict__ pre, int rows, float* __restrict__ partial) {
    constexpr int RPW = 32 / STR;                 // rows covered per wave load step
    __shared__ float ls[256], lq[256];
    const int wave = threadIdx.x >> 5;
    const int lane = threadIdx.x & 31;
    const int gw = blockIdx.x * 8 + wave;         // global wave id
    const int nw = gridDim.x * 8;
    const int c  = lane % STR;
    const int ro = lane / STR;
    float s = 0.f, q = 0.f;
    for (int r = gw * RPW + ro; r < rows; r += nw * RPW) {
        float v = pre[(size_t)r * STR + c];       // fully coalesced across the wave
        s += v; q += v * v;
    }
    ls[threadIdx.x] = s; lq[threadIdx.x] = q;
    __syncthreads();
    if (ro == 0) {
#pragma unroll
        for (int j = 1; j < RPW; ++j) { s += ls[threadIdx.x + j * STR]; q += lq[threadIdx.x + j * STR]; }
        partial[(size_t)gw * 2 * STR + c]       = s;
        partial[(size_t)gw * 2 * STR + STR + c] = q;
    }
}

template <int STR>
__global__ void bn_stats_reduce_kernel(const float* __restrict__ partial, int nw,
                                       float* __restrict__ stats) {
    int t = threadIdx.x;
    if (t >= 2 * STR) return;
    float s = 0.f;
    for (int w = 0; w < nw; ++w) s += partial[(size_t)w * 2 * STR + t];  // fixed order
    int c = (t < STR) ? t : t - STR;
    if (t < STR) stats[2 * c] = s; else stats[2 * c + 1] = s;
}

__global__ void bn_apply_kernel(const float* __restrict__ pre, int rows, int instride,
                                int C, int outstride, const float* __restrict__ stats,
                                const float* __restrict__ gamma, const float* __restrict__ beta,
                                float* __restrict__ out) {
    int t = blockIdx.x * blockDim.x + threadIdx.x;
    if (t >= rows * C) return;
    int c = t % C, r = t / C;
    float inv = 1.f / (float)rows;
    float mu  = stats[2 * c] * inv;
    float var = stats[2 * c + 1] * inv - mu * mu;
    float sc  = gamma[c] * rsqrtf(var + EPS);
    float v   = (pre[(size_t)r * instride + c] - mu) * sc + beta[c];
    out[(size_t)r * outstride + c] = leaky(v);
}

// ---------------------------------------------------------------------------
// Host orchestration
// ---------------------------------------------------------------------------
static inline char* carve(char*& p, size_t bytes) {
    char* r = p;
    p += (bytes + 255) & ~(size_t)255;
    return r;
}

extern "C" void kernel_launch(void* const* d_in, const int* in_sizes, int n_in,
                              void* d_out, int out_size, void* d_ws, size_t ws_size,
                              hipStream_t stream) {
    (void)in_sizes; (void)n_in; (void)out_size; (void)ws_size;
    const float* x       = (const float*)d_in[0];   // (1024,128)
    const float* W_dense = (const float*)d_in[1];   // (128,6400)
    const float* b_dense = (const float*)d_in[2];   // (6400,)
    const float* We0     = (const float*)d_in[3];   // (64,2048)
    const float* We1     = (const float*)d_in[4];   // (32,96)
    const float* be0     = (const float*)d_in[5];   // (2048,)
    const float* be1     = (const float*)d_in[6];   // (96,)
    const float* root0   = (const float*)d_in[7];   // (64,32)
    const float* root1   = (const float*)d_in[8];   // (32,3)
    const float* rb0     = (const float*)d_in[9];   // (32,)
    const float* rb1     = (const float*)d_in[10];  // (3,)
    const float* g0      = (const float*)d_in[11];
    const float* g1      = (const float*)d_in[12];
    const float* bt0     = (const float*)d_in[13];
    const float* bt1     = (const float*)d_in[14];
    float* out = (float*)d_out;                     // (1024,100,3)

    char* p = (char*)d_ws;
    _Float16* xh   = (_Float16*)carve(p, (size_t)BATCH * 128 * 2);
    _Float16* Bdp  = (_Float16*)carve(p, (size_t)128 * 6400 * 2);
    _Float16* Wf1  = (_Float16*)carve(p, (size_t)4096 * 32 * 2);
    _Float16* Wf2  = (_Float16*)carve(p, (size_t)1024 * 16 * 2);
    float* h0      = (float*)carve(p, (size_t)NODES * 64 * 4);
    int*   idx1    = (int*)  carve(p, (size_t)NODES * 4);
    float* xj1     = (float*)carve(p, (size_t)NODES * 64 * 4);
    float* pre1    = (float*)carve(p, (size_t)NODES * 32 * 4);
    float* h1      = (float*)carve(p, (size_t)NODES * 32 * 4);
    int*   idx2    = (int*)  carve(p, (size_t)NODES * 4);
    float* xj2     = (float*)carve(p, (size_t)NODES * 32 * 4);
    float* pre2    = (float*)carve(p, (size_t)NODES * 16 * 4);
    float* partial = (float*)carve(p, (size_t)STAT_WAVES * 64 * 4);
    float* stats1  = (float*)carve(p, 2 * 32 * 4);
    float* stats2  = (float*)carve(p, 2 * 16 * 4);

    // 0) precision conversions / fragment-major weight repacks
    cvt_f16_kernel<<<(BATCH * 128 + 255) / 256, 256, 0, stream>>>(x, xh, BATCH * 128);
    build_bdense_kernel<<<(128 * 6400 + 255) / 256, 256, 0, stream>>>(W_dense, Bdp);
    build_wflat1_kernel<<<(131072 + 255) / 256, 256, 0, stream>>>(We0, Wf1);
    build_wflat2_kernel<<<(16384 + 255) / 256, 256, 0, stream>>>(We1, Wf2);

    // 1) dense layer -> h0 (== node features (102400, 64); reshape is a layout no-op)
    dense_wmma_kernel<<<800, 256, 0, stream>>>(xh, Bdp, b_dense, h0);

    // 2) layer 1: knn -> gather -> root/bias base -> fused outer-product WMMA msg
    knn1_kernel<64><<<BATCH, 128, 0, stream>>>(h0, idx1);
    gather_kernel<64><<<(NODES * 16 + 255) / 256, 256, 0, stream>>>(h0, idx1, xj1);
    base_kernel<64, 32, 32><<<(NODES * 32 + 255) / 256, 256, 0, stream>>>(h0, xj1, root0, rb0, be0, pre1);
    msg_wmma_kernel<64, 2><<<NODES / 16 / 8, 256, 0, stream>>>(h0, xj1, Wf1, pre1);
    bn_stats_part_kernel<32><<<STAT_WAVES / 8, 256, 0, stream>>>(pre1, NODES, partial);
    bn_stats_reduce_kernel<32><<<1, 64, 0, stream>>>(partial, STAT_WAVES, stats1);
    bn_apply_kernel<<<(NODES * 32 + 255) / 256, 256, 0, stream>>>(pre1, NODES, 32, 32, 32,
                                                                  stats1, g0, bt0, h1);

    // 3) layer 2 (dout=3 padded to one 16-wide tile; extra cols carry zeros)
    knn1_kernel<32><<<BATCH, 128, 0, stream>>>(h1, idx2);
    gather_kernel<32><<<(NODES * 8 + 255) / 256, 256, 0, stream>>>(h1, idx2, xj2);
    base_kernel<32, 3, 16><<<(NODES * 16 + 255) / 256, 256, 0, stream>>>(h1, xj2, root1, rb1, be1, pre2);
    msg_wmma_kernel<32, 1><<<NODES / 16 / 8, 256, 0, stream>>>(h1, xj2, Wf2, pre2);
    bn_stats_part_kernel<16><<<STAT_WAVES / 8, 256, 0, stream>>>(pre2, NODES, partial);
    bn_stats_reduce_kernel<16><<<1, 32, 0, stream>>>(partial, STAT_WAVES, stats2);
    bn_apply_kernel<<<(NODES * 3 + 255) / 256, 256, 0, stream>>>(pre2, NODES, 16, 3, 3,
                                                                 stats2, g1, bt1, out);
}
